// EncoderLayer_85100482003345
// MI455X (gfx1250) — matrix operand order repaired and checked by reading, not compiled, not run on hardware
//
#include <hip/hip_runtime.h>
#include <hip/hip_bf16.h>
#include <cstdint>
#include <cstddef>

// ---------------------------------------------------------------------------
// Encoder layer (pre-norm MHA + FFN) for MI455X (gfx1250, wave32, WMMA).
// GEMMs: v_wmma_f32_16x16x32_bf16 (bf16 in, f32 accumulate).
// Weight tiles staged block-wide into LDS via global_load_async_to_lds_b128
// (ASYNCcnt path), double-buffered and overlapped with WMMA.
// ---------------------------------------------------------------------------

#define NB   2
#define NS   2048
#define ND   768
#define NH   12
#define NDK  64
#define NDFF 2048
#define NM   (NB * NS)   // 4096 token rows

typedef __attribute__((ext_vector_type(16))) __bf16          v16bf;
typedef __attribute__((ext_vector_type(8)))  float           v8f;
typedef __attribute__((ext_vector_type(8)))  unsigned short  u16x8;

// float -> bf16 round-to-nearest-even
__device__ __forceinline__ unsigned short f2bf(float f) {
    unsigned u = __builtin_bit_cast(unsigned, f);
    u += 0x7fffu + ((u >> 16) & 1u);
    return (unsigned short)(u >> 16);
}

__device__ __forceinline__ u16x8 ld8(const unsigned short* p) {
    return *(const u16x8*)p;
}

__device__ __forceinline__ v16bf mk16(u16x8 lo, u16x8 hi) {
    v16bf r;
#pragma unroll
    for (int i = 0; i < 8; ++i) {
        r[i]     = __builtin_bit_cast(__bf16, (unsigned short)lo[i]);
        r[i + 8] = __builtin_bit_cast(__bf16, (unsigned short)hi[i]);
    }
    return r;
}

__device__ __forceinline__ v8f wmma_bf16(v16bf a, v16bf b, v8f c) {
    return __builtin_amdgcn_wmma_f32_16x16x32_bf16(
        /*neg_a=*/false, a, /*neg_b=*/false, b,
        /*c_mod=*/(short)0, c, /*reuse_a=*/false, /*reuse_b=*/false);
}

// Async copy of 16 bytes global -> LDS, tracked by ASYNCcnt.
__device__ __forceinline__ void async_ld16(unsigned lds_off, const void* g) {
    asm volatile("global_load_async_to_lds_b128 %0, %1, off"
                 :: "v"(lds_off), "v"((unsigned long long)(uintptr_t)g)
                 : "memory");
}
__device__ __forceinline__ void wait_async0() {
    asm volatile("s_wait_asynccnt 0x0" ::: "memory");
}

// ---------------------------------------------------------------------------
// fp32 -> bf16 convert (weights)
// ---------------------------------------------------------------------------
__global__ void cvt_bf16_kernel(const float* __restrict__ in,
                                unsigned short* __restrict__ out, int n) {
    int i = blockIdx.x * blockDim.x + threadIdx.x;
    if (i < n) out[i] = f2bf(in[i]);
}

// ---------------------------------------------------------------------------
// LayerNorm (mean, std with ddof=1, eps added to std), output bf16.
// One block (256 threads) per row of D=768.
// ---------------------------------------------------------------------------
__global__ void layernorm_bf16_kernel(const float* __restrict__ x,
                                      const float* __restrict__ alpha,
                                      const float* __restrict__ beta,
                                      unsigned short* __restrict__ out) {
    __shared__ float ssum[256];
    __shared__ float ssq[256];
    const int row  = blockIdx.x;
    const int tid  = threadIdx.x;
    const float* xr = x + (size_t)row * ND;

    float s = 0.f, sq = 0.f;
#pragma unroll
    for (int i = 0; i < ND / 256; ++i) {
        float v = xr[tid + i * 256];
        s += v; sq += v * v;
    }
    ssum[tid] = s; ssq[tid] = sq;
    __syncthreads();
    for (int off = 128; off > 0; off >>= 1) {
        if (tid < off) { ssum[tid] += ssum[tid + off]; ssq[tid] += ssq[tid + off]; }
        __syncthreads();
    }
    const float mu  = ssum[0] * (1.f / ND);
    const float var = (ssq[0] - (float)ND * mu * mu) * (1.f / (ND - 1));
    const float den = 1.f / (sqrtf(fmaxf(var, 0.f)) + 1e-6f);

    unsigned short* orow = out + (size_t)row * ND;
#pragma unroll
    for (int i = 0; i < ND / 256; ++i) {
        int d = tid + i * 256;
        float v = alpha[d] * ((xr[d] - mu) * den) + beta[d];
        orow[d] = f2bf(v);
    }
}

// ---------------------------------------------------------------------------
// WMMA GEMM: C[M,N] = A[M,K] (bf16, row-major) x W[N,K]^T (bf16, row-major)
//            + bias[N], with mode-specific epilogue.
// Block: 256 threads = 8 waves; each wave computes one 16x16 tile over full K.
// Block tile: 128 rows x 16 cols.  The 16x32 W slab is block-uniform, so it
// is staged into LDS once per K-step with global_load_async_to_lds_b128
// (double-buffered, overlapped with the current WMMA).
// MODE:
//   0: out f32 = acc + bias + residual          (output proj / FFN2)
//   1: out bf16 = relu(acc + bias)              (FFN1)
//   2: out bf16 scattered to [B,H,S,DK]         (Q, K)
//   3: out bf16 scattered to [B,H,DK,S]         (V, pre-transposed)
// ---------------------------------------------------------------------------
template <int MODE>
__global__ void wmma_gemm_kernel(const unsigned short* __restrict__ A,
                                 const unsigned short* __restrict__ W,
                                 const float* __restrict__ bias,
                                 const float* __restrict__ res,
                                 void* __restrict__ outv,
                                 int M, int N, int K) {
    __shared__ unsigned short wtile[2][16 * 32];   // 2 x 1KB W slabs

    const int tid  = threadIdx.x;
    const int wave = tid >> 5;
    const int lane = tid & 31;
    const int half = lane >> 4;     // 0: lanes 0-15, 1: lanes 16-31
    const int ln   = lane & 15;

    const int row0 = blockIdx.y * 128 + wave * 16;
    const int col0 = blockIdx.x * 16;

    // A-operand: lane holds matrix row (ln); K split 8+8 per ISA layout.
    const unsigned short* ap = A + (size_t)(row0 + ln) * K + half * 8;

    const unsigned ldsbase = (unsigned)(uintptr_t)(&wtile[0][0]);

    // 64 lanes (waves 0-1) each async-copy 16B of the 1KB W slab.
    auto issue_w = [&](int buf, int k) {
        if (tid < 64) {
            const int row = tid >> 2, chunk = tid & 3;
            const unsigned short* g =
                W + (size_t)(col0 + row) * K + k + chunk * 8;
            async_ld16(ldsbase + (unsigned)buf * (16 * 32 * 2) + tid * 16, g);
        }
    };

    int cur = 0;
    issue_w(0, 0);
    wait_async0();
    __syncthreads();

    v8f acc = {};
    for (int k = 0; k < K; k += 32) {
        if (k + 32 < K) issue_w(cur ^ 1, k + 32);   // prefetch next slab
        if (k + 64 < K) __builtin_prefetch(ap + k + 64, 0, 0);

        v16bf av = mk16(ld8(ap + k), ld8(ap + k + 16));
        // B-operand from LDS: lane holds matrix column (ln); K split 8+8 per ISA layout.
        const unsigned short* bp = &wtile[cur][ln * 32 + half * 8];
        v16bf wv = mk16(ld8(bp), ld8(bp + 16));
        acc = wmma_bf16(av, wv, acc);

        wait_async0();        // next slab landed (no-op for non-loader waves)
        __syncthreads();
        cur ^= 1;
    }

    const int   n  = col0 + ln;
    const float bn = bias[n];

#pragma unroll
    for (int r = 0; r < 8; ++r) {
        const int   m = row0 + r + half * 8;
        const float v = acc[r] + bn;
        if (MODE == 0) {
            float* of = (float*)outv;
            size_t idx = (size_t)m * N + n;
            of[idx] = v + res[idx];
        } else if (MODE == 1) {
            unsigned short* ob = (unsigned short*)outv;
            ob[(size_t)m * N + n] = f2bf(fmaxf(v, 0.f));
        } else if (MODE == 2) {
            // m = b*S + s ; n = h*DK + dk  ->  [B,H,S,DK]
            unsigned short* ob = (unsigned short*)outv;
            int b = m >> 11, s = m & (NS - 1);
            int h = n >> 6,  dk = n & (NDK - 1);
            ob[(((size_t)b * NH + h) * NS + s) * NDK + dk] = f2bf(v);
        } else {
            // [B,H,DK,S]  (V transposed so PV B-operand is K-contiguous)
            unsigned short* ob = (unsigned short*)outv;
            int b = m >> 11, s = m & (NS - 1);
            int h = n >> 6,  dk = n & (NDK - 1);
            ob[(((size_t)b * NH + h) * NDK + dk) * NS + s] = f2bf(v);
        }
    }
}

// ---------------------------------------------------------------------------
// Flash attention, one wave = one 16-row query tile, online softmax over
// 32-key tiles. QK^T and PV via WMMA bf16. P tile converted C-layout ->
// A-operand layout through wave-private LDS (DS ops are in-order per wave).
// Grid: (S/128, B*H); block 256 (8 waves).
// Q,K: [B,H,S,DK] bf16 ; Vt: [B,H,DK,S] bf16 ; out: [B*S, D] bf16 (concat)
// ---------------------------------------------------------------------------
__global__ void attention_kernel(const unsigned short* __restrict__ Q,
                                 const unsigned short* __restrict__ Km,
                                 const unsigned short* __restrict__ Vt,
                                 const int* __restrict__ mask,
                                 unsigned short* __restrict__ out) {
    __shared__ unsigned short pbuf[8][16 * 32];

    const int bh   = blockIdx.y;
    const int b    = bh / NH;
    const int h    = bh - b * NH;
    const int wave = threadIdx.x >> 5;
    const int lane = threadIdx.x & 31;
    const int half = lane >> 4;
    const int ln   = lane & 15;
    const int qbase = blockIdx.x * 128 + wave * 16;

    const unsigned short* qp = Q  + (size_t)bh * NS * NDK;
    const unsigned short* kp = Km + (size_t)bh * NS * NDK;
    const unsigned short* vp = Vt + (size_t)bh * NDK * NS;
    unsigned short* pb = &pbuf[wave][0];

    // Q A-operands for the two 32-wide d chunks (DK = 64).
    v16bf aq[2];
#pragma unroll
    for (int c = 0; c < 2; ++c) {
        const unsigned short* p = qp + (size_t)(qbase + ln) * NDK + c * 32 + half * 8;
        aq[c] = mk16(ld8(p), ld8(p + 16));
    }

    v8f   oacc[4] = {};                      // 16 x 64 output tile (4 C tiles)
    float mrow[8], lrow[8];
#pragma unroll
    for (int r = 0; r < 8; ++r) { mrow[r] = -1e30f; lrow[r] = 0.f; }

    for (int kb = 0; kb < NS; kb += 32) {
        // ---- scores: two 16x16 tiles (keys kb..kb+15, kb+16..kb+31) ----
        v8f s0 = {}, s1 = {};
#pragma unroll
        for (int c = 0; c < 2; ++c) {
            const unsigned short* p0 = kp + (size_t)(kb + ln) * NDK       + c * 32 + half * 8;
            const unsigned short* p1 = kp + (size_t)(kb + 16 + ln) * NDK  + c * 32 + half * 8;
            s0 = wmma_bf16(aq[c], mk16(ld8(p0), ld8(p0 + 16)), s0);
            s1 = wmma_bf16(aq[c], mk16(ld8(p1), ld8(p1 + 16)), s1);
        }
        const int ms0 = mask[b * NS + kb + ln];
        const int ms1 = mask[b * NS + kb + 16 + ln];
#pragma unroll
        for (int r = 0; r < 8; ++r) {
            s0[r] = ms0 ? s0[r] * 0.125f : -1e9f;   // 1/sqrt(64)
            s1[r] = ms1 ? s1[r] * 0.125f : -1e9f;
        }

        // ---- online softmax: per-row max/sum across the 16 lanes of half ----
        float tm[8];
#pragma unroll
        for (int r = 0; r < 8; ++r) tm[r] = fmaxf(s0[r], s1[r]);
#pragma unroll
        for (int xm = 1; xm < 16; xm <<= 1)
#pragma unroll
            for (int r = 0; r < 8; ++r)
                tm[r] = fmaxf(tm[r], __shfl_xor(tm[r], xm, 32));

        float ts[8];
#pragma unroll
        for (int r = 0; r < 8; ++r) {
            const float nm = fmaxf(mrow[r], tm[r]);
            const float al = __expf(mrow[r] - nm);
            mrow[r] = nm;
            const float p0 = __expf(s0[r] - nm);
            const float p1 = __expf(s1[r] - nm);
            ts[r] = p0 + p1;
            lrow[r] = lrow[r] * al;
            oacc[0][r] *= al; oacc[1][r] *= al; oacc[2][r] *= al; oacc[3][r] *= al;
            const int mm = r + half * 8;
            pb[mm * 32 + ln]      = f2bf(p0);
            pb[mm * 32 + 16 + ln] = f2bf(p1);
        }
#pragma unroll
        for (int xm = 1; xm < 16; xm <<= 1)
#pragma unroll
            for (int r = 0; r < 8; ++r)
                ts[r] += __shfl_xor(ts[r], xm, 32);
#pragma unroll
        for (int r = 0; r < 8; ++r) lrow[r] += ts[r];

        // ---- P (A-operand) from LDS, then PV over 4 d tiles ----
        const unsigned short* pr = pb + ln * 32 + half * 8;
        v16bf pa = mk16(ld8(pr), ld8(pr + 16));
#pragma unroll
        for (int dt = 0; dt < 4; ++dt) {
            const unsigned short* vr = vp + (size_t)(dt * 16 + ln) * NS + kb + half * 8;
            oacc[dt] = wmma_bf16(pa, mk16(ld8(vr), ld8(vr + 16)), oacc[dt]);
        }
    }

    // ---- normalize and store concat [B*S, D] bf16 ----
#pragma unroll
    for (int r = 0; r < 8; ++r) {
        const float inv = 1.f / lrow[r];
        oacc[0][r] *= inv; oacc[1][r] *= inv; oacc[2][r] *= inv; oacc[3][r] *= inv;
    }
#pragma unroll
    for (int dt = 0; dt < 4; ++dt)
#pragma unroll
        for (int r = 0; r < 8; ++r) {
            const int mm   = r + half * 8;
            const int srow = qbase + mm;
            out[((size_t)(b * NS + srow)) * ND + h * NDK + dt * 16 + ln] =
                f2bf(oacc[dt][r]);
        }
}

// ---------------------------------------------------------------------------
// Host launcher
// ---------------------------------------------------------------------------
extern "C" void kernel_launch(void* const* d_in, const int* in_sizes, int n_in,
                              void* d_out, int out_size, void* d_ws, size_t ws_size,
                              hipStream_t stream) {
    (void)in_sizes; (void)n_in; (void)out_size; (void)ws_size;

    const float* x     = (const float*)d_in[0];
    const int*   mask  = (const int*)d_in[1];
    const float* Wq    = (const float*)d_in[2];
    const float* bq    = (const float*)d_in[3];
    const float* Wk    = (const float*)d_in[4];
    const float* bk    = (const float*)d_in[5];
    const float* Wv    = (const float*)d_in[6];
    const float* bv    = (const float*)d_in[7];
    const float* Wo    = (const float*)d_in[8];
    const float* bo    = (const float*)d_in[9];
    const float* W1    = (const float*)d_in[10];
    const float* b1    = (const float*)d_in[11];
    const float* W2    = (const float*)d_in[12];
    const float* b2    = (const float*)d_in[13];
    const float* alpha = (const float*)d_in[14];
    const float* beta  = (const float*)d_in[15];
    float* outp = (float*)d_out;

    // ---- workspace carve (~72 MB) ----
    char*  base = (char*)d_ws;
    size_t off  = 0;
    auto carve = [&](size_t bytes) -> void* {
        off = (off + 255) & ~(size_t)255;
        void* p = base + off;
        off += bytes;
        return p;
    };
    unsigned short* nbf  = (unsigned short*)carve((size_t)NM * ND * 2);
    unsigned short* wqb  = (unsigned short*)carve((size_t)ND * ND * 2);
    unsigned short* wkb  = (unsigned short*)carve((size_t)ND * ND * 2);
    unsigned short* wvb  = (unsigned short*)carve((size_t)ND * ND * 2);
    unsigned short* wob  = (unsigned short*)carve((size_t)ND * ND * 2);
    unsigned short* w1b  = (unsigned short*)carve((size_t)NDFF * ND * 2);
    unsigned short* w2b  = (unsigned short*)carve((size_t)ND * NDFF * 2);
    unsigned short* qb   = (unsigned short*)carve((size_t)NM * ND * 2);
    unsigned short* kbuf = (unsigned short*)carve((size_t)NM * ND * 2);
    unsigned short* vtb  = (unsigned short*)carve((size_t)NM * ND * 2);
    unsigned short* catb = (unsigned short*)carve((size_t)NM * ND * 2);
    float*          x1   = (float*)carve((size_t)NM * ND * 4);
    unsigned short* hbf  = (unsigned short*)carve((size_t)NM * NDFF * 2);

    // ---- weights -> bf16 ----
    auto cvt = [&](const float* src, unsigned short* dst, int n) {
        cvt_bf16_kernel<<<(n + 255) / 256, 256, 0, stream>>>(src, dst, n);
    };
    cvt(Wq, wqb, ND * ND);
    cvt(Wk, wkb, ND * ND);
    cvt(Wv, wvb, ND * ND);
    cvt(Wo, wob, ND * ND);
    cvt(W1, w1b, NDFF * ND);
    cvt(W2, w2b, ND * NDFF);

    const dim3 gD(ND / 16, NM / 128);     // N=768 GEMMs
    const dim3 gF(NDFF / 16, NM / 128);   // N=2048 GEMM

    // ---- pre-norm 1 ----
    layernorm_bf16_kernel<<<NM, 256, 0, stream>>>(x, alpha, beta, nbf);

    // ---- Q,K,V projections (scattered to attention layouts) ----
    wmma_gemm_kernel<2><<<gD, 256, 0, stream>>>(nbf, wqb, bq, nullptr, qb,   NM, ND, ND);
    wmma_gemm_kernel<2><<<gD, 256, 0, stream>>>(nbf, wkb, bk, nullptr, kbuf, NM, ND, ND);
    wmma_gemm_kernel<3><<<gD, 256, 0, stream>>>(nbf, wvb, bv, nullptr, vtb,  NM, ND, ND);

    // ---- attention ----
    attention_kernel<<<dim3(NS / 128, NB * NH), 256, 0, stream>>>(qb, kbuf, vtb, mask, catb);

    // ---- output projection + residual ----
    wmma_gemm_kernel<0><<<gD, 256, 0, stream>>>(catb, wob, bo, x, x1, NM, ND, ND);

    // ---- pre-norm 2 + FFN ----
    layernorm_bf16_kernel<<<NM, 256, 0, stream>>>(x1, alpha, beta, nbf);
    wmma_gemm_kernel<1><<<gF, 256, 0, stream>>>(nbf, w1b, b1, nullptr, hbf, NM, NDFF, ND);
    wmma_gemm_kernel<0><<<gD, 256, 0, stream>>>(hbf, w2b, b2, x1, outp, NM, ND, NDFF);
}
